// KL_distance_loss_33964601377289
// MI455X (gfx1250) — matrix operand order repaired and checked by reading, not compiled, not run on hardware
//
#include <hip/hip_runtime.h>
#include <hip/hip_bf16.h>

typedef __attribute__((ext_vector_type(16))) _Float16 v16h;
typedef __attribute__((ext_vector_type(8)))  float    v8f;

#define NROW     4096   // B*S*FEAT / 4096
#define ROWLEN_X 4096
#define ROWLEN_P 1024
#define LPATCH   64
#define IPATCH   64
#define JPATCH   16
#define LDST     68     // LDS row stride in halves (136B, conflict-free gathers)

__device__ __forceinline__ float wave_max(float v) {
    #pragma unroll
    for (int d = 16; d >= 1; d >>= 1) v = fmaxf(v, __shfl_xor(v, d, 32));
    return v;
}
__device__ __forceinline__ float wave_sum(float v) {
    #pragma unroll
    for (int d = 16; d >= 1; d >>= 1) v += __shfl_xor(v, d, 32);
    return v;
}

union HPack { _Float16 h[2]; unsigned u; };
union AFrag { unsigned u[8]; v16h h; };

__global__ __launch_bounds__(128) void kl_row_kernel(const float* __restrict__ x,
                                                     const float* __restrict__ p,
                                                     float* __restrict__ partial) {
    __shared__ __align__(16) _Float16 s_logp[IPATCH * LDST]; // [i][l] f16
    __shared__ __align__(16) _Float16 s_q[JPATCH * LDST];    // [j][l] f16
    __shared__ float s_negent[JPATCH];
    __shared__ float s_wsum[4];

    const int n    = blockIdx.x;
    const int tid  = threadIdx.x;
    const int wave = tid >> 5;
    const int lane = tid & 31;

    const float* xrow = x + (size_t)n * ROWLEN_X;
    const float* prow = p + (size_t)n * ROWLEN_P;

    // ---- Phase 1a: log-softmax of the 64 x-patches (16 per wave, wave-cooperative)
    #pragma unroll 4
    for (int t = 0; t < 16; ++t) {
        const int i = wave * 16 + t;
        const float2 ab = ((const float2*)(xrow + i * LPATCH))[lane]; // coalesced b64
        const float mx = wave_max(fmaxf(ab.x, ab.y));
        const float s  = wave_sum(__expf(ab.x - mx) + __expf(ab.y - mx));
        const float lz = mx + __logf(s);
        HPack pk;
        pk.h[0] = (_Float16)(ab.x - lz);
        pk.h[1] = (_Float16)(ab.y - lz);
        *(unsigned*)&s_logp[i * LDST + 2 * lane] = pk.u;
    }

    // ---- Phase 1b: log-softmax + q + neg-entropy of the 16 p-patches (4 per wave)
    #pragma unroll
    for (int t = 0; t < 4; ++t) {
        const int j = wave * 4 + t;
        const float2 ab = ((const float2*)(prow + j * LPATCH))[lane];
        const float mx = wave_max(fmaxf(ab.x, ab.y));
        const float s  = wave_sum(__expf(ab.x - mx) + __expf(ab.y - mx));
        const float lz = mx + __logf(s);
        const float lq0 = ab.x - lz, lq1 = ab.y - lz;
        const float q0 = __expf(lq0), q1 = __expf(lq1);
        HPack pk;
        pk.h[0] = (_Float16)q0;
        pk.h[1] = (_Float16)q1;
        *(unsigned*)&s_q[j * LDST + 2 * lane] = pk.u;
        const float ent = wave_sum(q0 * lq0 + q1 * lq1);
        if (lane == 0) s_negent[j] = ent;
    }
    __syncthreads();

    // ---- Phase 2: cross[i,j] = sum_l q[j,l] * log_p[i,l] via v_wmma_f32_16x16x32_f16
    // Wave w owns M-tile w (i = w*16 .. w*16+15); single N-tile (j=0..15); K=64 in 2 steps.
    const int m  = lane & 15;   // A: row M; B/C: column N
    const int hi = lane >> 4;

    v8f c = {};
    #pragma unroll
    for (int kt = 0; kt < 2; ++kt) {
        AFrag af;
        const _Float16* lpRow = &s_logp[(wave * 16 + m) * LDST];
        #pragma unroll
        for (int r = 0; r < 8; ++r) {
            af.u[r] = *(const unsigned*)&lpRow[kt * 32 + hi * 16 + 2 * r];
        }
        AFrag bf;
        const _Float16* qRow = &s_q[m * LDST];
        #pragma unroll
        for (int r = 0; r < 8; ++r) {
            bf.u[r] = *(const unsigned*)&qRow[kt * 32 + hi * 16 + 2 * r];
        }
        c = __builtin_amdgcn_wmma_f32_16x16x32_f16(
                /*neg_a=*/false, af.h, /*neg_b=*/false, bf.h,
                /*c_mod=*/(short)0, c, /*reuse_a=*/false, /*reuse_b=*/false);
    }

    // ---- Phase 3: kl = neg_ent[j] - cross; softmax over j (16 lanes of half-group);
    //      accumulate sum_{i,j} w*kl
    const float neg = s_negent[m];  // j == N == lane&15
    float acc = 0.f;
    #pragma unroll
    for (int v = 0; v < 8; ++v) {
        const float kl = neg - c[v];
        float mx = kl;
        #pragma unroll
        for (int d = 8; d >= 1; d >>= 1) mx = fmaxf(mx, __shfl_xor(mx, d, 32));
        const float e = __expf(kl - mx);
        float s = e;
        #pragma unroll
        for (int d = 8; d >= 1; d >>= 1) s += __shfl_xor(s, d, 32);
        acc += (e / s) * kl;   // each lane contributes its own (i,j) term
    }
    acc = wave_sum(acc);
    if (lane == 0) s_wsum[wave] = acc;
    __syncthreads();
    if (tid == 0) partial[n] = s_wsum[0] + s_wsum[1] + s_wsum[2] + s_wsum[3];
}

__global__ __launch_bounds__(256) void kl_reduce_kernel(const float* __restrict__ partial,
                                                        float* __restrict__ out) {
    __shared__ float s[256];
    float a = 0.f;
    for (int i = threadIdx.x; i < NROW; i += 256) a += partial[i];
    s[threadIdx.x] = a;
    __syncthreads();
    #pragma unroll
    for (int st = 128; st > 0; st >>= 1) {
        if (threadIdx.x < st) s[threadIdx.x] += s[threadIdx.x + st];
        __syncthreads();
    }
    if (threadIdx.x == 0) out[0] = s[0] * (1.0f / (float)NROW);
}

extern "C" void kernel_launch(void* const* d_in, const int* in_sizes, int n_in,
                              void* d_out, int out_size, void* d_ws, size_t ws_size,
                              hipStream_t stream) {
    const float* batch_x = (const float*)d_in[0];  // [32,4096,128] f32
    const float* pred    = (const float*)d_in[1];  // [32,1024,128] f32
    (void)in_sizes; (void)n_in; (void)out_size; (void)ws_size;
    float* partial = (float*)d_ws;                 // NROW floats of scratch
    float* out     = (float*)d_out;                // scalar f32 loss

    kl_row_kernel<<<NROW, 128, 0, stream>>>(batch_x, pred, partial);
    kl_reduce_kernel<<<1, 256, 0, stream>>>(partial, out);
}
